// FractionalLossFunction_54786602828152
// MI455X (gfx1250) — matrix operand order repaired and checked by reading, not compiled, not run on hardware
//
#include <hip/hip_runtime.h>

typedef __attribute__((ext_vector_type(8)))  float  v8f;
typedef __attribute__((ext_vector_type(8)))  __bf16 v8bf;
typedef __attribute__((ext_vector_type(16))) __bf16 v16bf;
typedef __attribute__((ext_vector_type(4)))  unsigned u32x4;
typedef __attribute__((ext_vector_type(4)))  int      i32x4;
typedef __attribute__((ext_vector_type(8)))  int      i32x8;

#define NROW 8192
#define NCOL 2048
#define TILE_B 128
#define TILE_N 128
#define TILE_K 32
#define LDK     40   // Toeplitz tile row stride (ushorts): conflict-free b128 reads
#define RSTRIDE 36   // raw f32 tile row stride (DWORDs): 32 + 4 TDM pad DWORDs
#define ALPHA  0.5f

// round-to-nearest-even f32 -> bf16 bit pattern (for weight table build)
__device__ __forceinline__ unsigned bf16_rne(float x) {
    unsigned b = __float_as_uint(x);
    return (b + 0x7FFFu + ((b >> 16) & 1u)) >> 16;
}

// Kernel 1: zero the loss accumulator; build packed split GL weights
// wp[k] = (bf16_hi(w'_k) << 16) | bf16_lo(w'_k),  w'_k = h^{-alpha} * w_k
__global__ void frac_init_weights(unsigned* __restrict__ wp, float* __restrict__ out)
{
    if (blockIdx.x == 0 && threadIdx.x == 0) {
        out[0] = 0.0f;
        float cur = sqrtf((float)(NCOL - 1));           // h^{-0.5} = sqrt(N-1)
        for (int k = 0; k < NCOL; ++k) {
            if (k) cur *= ((float)(k - 1) - ALPHA) / (float)k;
            const unsigned h  = bf16_rne(cur);
            const float    hf = __uint_as_float(h << 16);
            const unsigned l  = bf16_rne(cur - hf);     // remainder exact in f32
            wp[k] = (h << 16) | l;
        }
    }
}

// ---- Tensor Data Mover: async 2D tile (TILE_K x TILE_B f32) global -> LDS.
// D# per CDNA5 ISA 8.3/8.4. LDS pad: every 32 DWORDs (+4 DWORDs) -> 36-DW rows,
// which makes the per-lane b128 A-fragment reads bank-conflict-free.
__device__ __forceinline__ void tdm_load_x_tile(const float* gsrc, unsigned lds_off)
{
    u32x4 g0;
    const unsigned long long ga = (unsigned long long)(size_t)gsrc;
    g0.x = 1u;                                           // count=1 (valid user D#)
    g0.y = lds_off;                                      // LDS byte address
    g0.z = (unsigned)(ga & 0xFFFFFFFFull);               // global_addr[31:0]
    g0.w = (unsigned)((ga >> 32) & 0x01FFFFFFull) | (2u << 30);  // [56:32] | type=2

    // data_size=4B(2) | pad_enable | pad_interval=4 (32 DW) | pad_amount=3 (4 DW)
    const unsigned w0 = (2u << 16) | (1u << 20) | (4u << 22) | (3u << 25);
    i32x8 g1;
    g1[0] = (int)w0;
    g1[1] = (int)((NCOL & 0xFFFFu) << 16);                    // tensor_dim0[15:0]
    g1[2] = (int)((NCOL >> 16) | ((NROW & 0xFFFFu) << 16));   // dim0 hi | dim1 lo
    g1[3] = (int)((NROW >> 16) | ((unsigned)TILE_K << 16));   // dim1 hi | tile_dim0
    g1[4] = (int)TILE_B;                                      // tile_dim1 | tile_dim2=0
    g1[5] = (int)NCOL;                                        // tensor_dim0_stride lo
    g1[6] = 0;                                                // stride hi | dim1_stride
    g1[7] = 0;
    const i32x4 z4 = {0, 0, 0, 0};
#if __clang_major__ >= 23
    const i32x8 z8 = {0, 0, 0, 0, 0, 0, 0, 0};
    __builtin_amdgcn_tensor_load_to_lds(g0, g1, z4, z4, z8, 0);
#else
    __builtin_amdgcn_tensor_load_to_lds(g0, g1, z4, z4, 0);
#endif
}

// split 8 f32 into bf16 hi/lo fragment halves, phase-ordered so the SLP
// vectorizer can pair both hi and lo casts into v_cvt_pk_bf16_f32 and the
// residuals into v_fma_mix_f32_bf16.
__device__ __forceinline__ void split8(const float4 f0, const float4 f1,
                                       v16bf& hi, v16bf& lo, int base)
{
    float a[8] = {f0.x, f0.y, f0.z, f0.w, f1.x, f1.y, f1.z, f1.w};
    __bf16 h[8];
#pragma unroll
    for (int e = 0; e < 8; ++e) h[e] = (__bf16)a[e];
#pragma unroll
    for (int e = 0; e < 8; ++e) hi[base + e] = h[e];
    float r[8];
#pragma unroll
    for (int e = 0; e < 8; ++e) r[e] = a[e] - (float)h[e];
#pragma unroll
    for (int e = 0; e < 8; ++e) lo[base + e] = (__bf16)r[e];
}

// Kernel 2: fused mean((X @ T^T - targets)^2), T lower-triangular Toeplitz.
// bf16x3 split GEMM on v_wmma_f32_16x16x32_bf16; X tiles staged by the TDM
// (async, double-buffered); A-fragments converted f32->bf16 hi/lo in registers.
__global__ __launch_bounds__(256) void frac_gemm_loss(
    const float*    __restrict__ X,
    const float*    __restrict__ Tg,
    const unsigned* __restrict__ wp,
    float*          __restrict__ out)
{
    __shared__ float          Xraw[2][TILE_B * RSTRIDE]; // TDM-staged f32, 2x18KB
    __shared__ unsigned short BhT[TILE_N * LDK];         // Toeplitz hi, [n][k]
    __shared__ unsigned short BlT[TILE_N * LDK];         // Toeplitz lo
    __shared__ float red[8];

    const int tid  = threadIdx.x;
    const int lane = tid & 31;
    const int wv   = tid >> 5;
    const int l16  = lane & 15;
    const int hsel = lane >> 4;          // 0: lanes 0-15, 1: lanes 16-31
    const int waveM = wv & 3;            // 4 wave-rows of 32 output rows
    const int waveN = wv >> 2;           // 2 wave-cols of 64 output cols

    const int n0 = blockIdx.x * TILE_N;
    const int b0 = blockIdx.y * TILE_B;

    const v8f vzero = {0.f, 0.f, 0.f, 0.f, 0.f, 0.f, 0.f, 0.f};
    v8f acc[2][4];
#pragma unroll
    for (int mi = 0; mi < 2; ++mi)
#pragma unroll
        for (int ni = 0; ni < 4; ++ni)
            acc[mi][ni] = vzero;

    // lower-triangular: columns <= n0+127 need only m <= n0+127
    const int ktmax = NCOL / TILE_K;
    int ktiles = (n0 + TILE_N) / TILE_K;
    ktiles = (ktiles < ktmax) ? ktiles : ktmax;

    const int rowA = waveM * 32 + l16;
    const int colB = waveN * 64 + l16;

    unsigned* BhU = (unsigned*)BhT;
    unsigned* BlU = (unsigned*)BlT;
    const unsigned raw_off0 = (unsigned)(size_t)&Xraw[0][0];
    const unsigned raw_off1 = (unsigned)(size_t)&Xraw[1][0];

    // prologue: DMA first X tile
    if (wv == 0)
        tdm_load_x_tile(&X[(size_t)b0 * NCOL], raw_off0);

    for (int kt = 0; kt < ktiles; ++kt) {
        const int m0  = kt * TILE_K;
        const int cur = kt & 1;

        if (wv == 0) __builtin_amdgcn_s_wait_tensorcnt(0);
        __syncthreads();   // raw[cur] published; prior iteration's LDS reads done

        // kick off DMA for the next X tile while we compute this one
        if (wv == 0 && (kt + 1) < ktiles)
            tdm_load_x_tile(&X[(size_t)b0 * NCOL + (size_t)(m0 + TILE_K)],
                            ((kt + 1) & 1) ? raw_off1 : raw_off0);

        // ---- stage Toeplitz tile: B[k][n] = w[n0 - m0 + n - k], stored [n][k].
        // Window is [D0-31, D0+127]; upper end <= n0+127 <= 2047 always, so only
        // the low-end guard is ever needed (last 4 K-iterations of each block).
        const int D0 = n0 - m0;
        if (D0 >= TILE_K - 1) {
#pragma unroll
            for (int p = 0; p < 8; ++p) {
                const int i  = tid + p * 256;
                const int n  = i >> 4;
                const int kp = i & 15;
                const int j0 = D0 + n - 2 * kp;
                const unsigned e0 = wp[j0];
                const unsigned e1 = wp[j0 - 1];
                const int o = n * (LDK / 2) + kp;
                BhU[o] = (e0 >> 16)     | (e1 & 0xFFFF0000u);
                BlU[o] = (e0 & 0xFFFFu) | (e1 << 16);
            }
        } else {
#pragma unroll
            for (int p = 0; p < 8; ++p) {
                const int i  = tid + p * 256;
                const int n  = i >> 4;
                const int kp = i & 15;
                const int j0 = D0 + n - 2 * kp;
                const int j1 = j0 - 1;
                const unsigned e0 = (j0 >= 0) ? wp[j0] : 0u;
                const unsigned e1 = (j1 >= 0) ? wp[j1] : 0u;
                const int o = n * (LDK / 2) + kp;
                BhU[o] = (e0 >> 16)     | (e1 & 0xFFFF0000u);
                BlU[o] = (e0 & 0xFFFFu) | (e1 << 16);
            }
        }
        __syncthreads();   // Toeplitz tile ready (raw tile already ready)

        // ---- A fragments straight from padded raw tile, split in registers.
        // A (16x32 bf16): lanes 0-15 hold K 0-7 & 16-23; lanes 16-31 K 8-15 & 24-31
        v16bf Ah[2], Al[2];
#pragma unroll
        for (int mi = 0; mi < 2; ++mi) {
            const float* rb = &Xraw[cur][(rowA + mi * 16) * RSTRIDE];
            const float4 f0 = *reinterpret_cast<const float4*>(&rb[8 * hsel]);
            const float4 f1 = *reinterpret_cast<const float4*>(&rb[8 * hsel + 4]);
            const float4 f2 = *reinterpret_cast<const float4*>(&rb[16 + 8 * hsel]);
            const float4 f3 = *reinterpret_cast<const float4*>(&rb[16 + 8 * hsel + 4]);
            split8(f0, f1, Ah[mi], Al[mi], 0);
            split8(f2, f3, Ah[mi], Al[mi], 8);
        }

        // B (32x16 bf16): lanes 0-15 hold K 0-7 & 16-23; lanes 16-31 K 8-15 & 24-31
#pragma unroll
        for (int ni = 0; ni < 4; ++ni) {
            const int n = colB + ni * 16;
            const v8bf h0 = *reinterpret_cast<const v8bf*>(&BhT[n * LDK + 8 * hsel]);
            const v8bf h1 = *reinterpret_cast<const v8bf*>(&BhT[n * LDK + 16 + 8 * hsel]);
            const v16bf Bh = __builtin_shufflevector(h0, h1, 0,1,2,3,4,5,6,7,8,9,10,11,12,13,14,15);
            const v8bf s0 = *reinterpret_cast<const v8bf*>(&BlT[n * LDK + 8 * hsel]);
            const v8bf s1 = *reinterpret_cast<const v8bf*>(&BlT[n * LDK + 16 + 8 * hsel]);
            const v16bf Bl = __builtin_shufflevector(s0, s1, 0,1,2,3,4,5,6,7,8,9,10,11,12,13,14,15);
#pragma unroll
            for (int mi = 0; mi < 2; ++mi) {
                acc[mi][ni] = __builtin_amdgcn_wmma_f32_16x16x32_bf16(
                    false, Ah[mi], false, Bh, (short)0, acc[mi][ni], false, false);
                acc[mi][ni] = __builtin_amdgcn_wmma_f32_16x16x32_bf16(
                    false, Ah[mi], false, Bl, (short)0, acc[mi][ni], false, false);
                acc[mi][ni] = __builtin_amdgcn_wmma_f32_16x16x32_bf16(
                    false, Al[mi], false, Bh, (short)0, acc[mi][ni], false, false);
            }
        }
    }

    // ---- fused loss epilogue: sum (y - t)^2 ; targets streamed non-temporally
    // (targets are read exactly once -> NT keeps X resident in the 192MB L2)
    float lsum = 0.0f;
#pragma unroll
    for (int mi = 0; mi < 2; ++mi) {
#pragma unroll
        for (int ni = 0; ni < 4; ++ni) {
            const int rowb = b0 + waveM * 32 + mi * 16 + 8 * hsel;
            const int col  = n0 + waveN * 64 + ni * 16 + l16;
            const float* tp = &Tg[(size_t)rowb * NCOL + col];
#pragma unroll
            for (int j = 0; j < 8; ++j) {
                const float t = __builtin_nontemporal_load(tp + (size_t)j * NCOL);
                const float d = acc[mi][ni][j] - t;
                lsum = fmaf(d, d, lsum);
            }
        }
    }

    // wave32 reduction -> LDS -> one atomic per block
#pragma unroll
    for (int off = 16; off > 0; off >>= 1)
        lsum += __shfl_xor(lsum, off, 32);
    if (lane == 0) red[wv] = lsum;
    __syncthreads();
    if (wv == 0) {
        float v = (lane < 8) ? red[lane] : 0.0f;
#pragma unroll
        for (int off = 4; off > 0; off >>= 1)
            v += __shfl_xor(v, off, 32);
        if (lane == 0) {
            const float inv = 1.0f / ((float)NROW * (float)NCOL);
            atomicAdd(out, v * inv);
        }
    }
}

extern "C" void kernel_launch(void* const* d_in, const int* in_sizes, int n_in,
                              void* d_out, int out_size, void* d_ws, size_t ws_size,
                              hipStream_t stream) {
    const float* X  = (const float*)d_in[0];   // predictions [8192, 2048] f32
    const float* Tg = (const float*)d_in[1];   // targets     [8192, 2048] f32
    float* out = (float*)d_out;                // scalar loss
    unsigned* wp = (unsigned*)d_ws;            // 2048 packed split GL weights

    frac_init_weights<<<1, 64, 0, stream>>>(wp, out);

    dim3 grid(NCOL / TILE_N, NROW / TILE_B);   // 16 x 64 tiles
    frac_gemm_loss<<<grid, 256, 0, stream>>>(X, Tg, wp, out);
}
